// SimpleGaussianHead_76416058131339
// MI455X (gfx1250) — matrix operand order repaired and checked by reading, not compiled, not run on hardware
//
#include <hip/hip_runtime.h>
#include <math.h>

// ---------------------------------------------------------------------------
// SimpleGaussianHead fused loss for gfx1250 (MI455X).
//   loss = mean_b[ logsumexp_c(logit) - logit_y ] + 0.1*mean(X^2)
//   logit[b,c] = (2*X.Wt - ||x||^2 - ||w||^2) / 1024
// GEMM (137 GFLOP) via v_wmma_f32_16x16x32_bf16; logits never hit HBM
// (online softmax in registers). W pre-converted to bf16 in workspace (L2
// resident, 16MB), X block staged as bf16 in LDS. Per-wave tile 32x32
// (4 accumulators) so each ds_load_b128 / global B frag feeds 2 WMMAs.
// ---------------------------------------------------------------------------

#define BATCH 8192
#define NCLS  8192
#define FEAT  1024
#define REG   0.1f

typedef __attribute__((ext_vector_type(16))) __bf16 v16bf;
typedef __attribute__((ext_vector_type(8)))  __bf16 v8bf;
typedef __attribute__((ext_vector_type(8)))  float  v8f;

__device__ __forceinline__ unsigned short f2b(float f) {
  // round-to-nearest-even f32 -> bf16 bits
  unsigned u = __float_as_uint(f);
  return (unsigned short)((u + 0x7FFFu + ((u >> 16) & 1u)) >> 16);
}

__device__ __forceinline__ float blockReduceSum(float v) {
  __shared__ float red[8];
  v += __shfl_xor(v, 1, 32);
  v += __shfl_xor(v, 2, 32);
  v += __shfl_xor(v, 4, 32);
  v += __shfl_xor(v, 8, 32);
  v += __shfl_xor(v, 16, 32);
  int lane = threadIdx.x & 31, wv = threadIdx.x >> 5;
  if (lane == 0) red[wv] = v;
  __syncthreads();
  if (wv == 0) {
    v = (lane < (int)(blockDim.x >> 5)) ? red[lane] : 0.0f;
    v += __shfl_xor(v, 1, 32);
    v += __shfl_xor(v, 2, 32);
    v += __shfl_xor(v, 4, 32);
  }
  return v; // valid on wave 0
}

// --- Prepass: row squared norms for X and W, W -> bf16, 0.1*mean(X^2) term --
__global__ void sgh_prep_kernel(const float* __restrict__ X,
                                const float* __restrict__ W,
                                float* __restrict__ x2, float* __restrict__ w2,
                                unsigned short* __restrict__ wbf,
                                float* __restrict__ out) {
  const int row = blockIdx.x;           // [0, BATCH+NCLS)
  const bool isX = row < BATCH;
  const float* src = isX ? (X + (size_t)row * FEAT)
                         : (W + (size_t)(row - BATCH) * FEAT);
  float s = 0.0f;
  const int k = threadIdx.x * 4;        // 256 threads * 4 = 1024 = FEAT
  float4 v = *reinterpret_cast<const float4*>(src + k);
  s = v.x * v.x + v.y * v.y + v.z * v.z + v.w * v.w;
  if (!isX) {
    unsigned short* p = wbf + (size_t)(row - BATCH) * FEAT + k;
    *reinterpret_cast<ushort4*>(p) =
        make_ushort4(f2b(v.x), f2b(v.y), f2b(v.z), f2b(v.w));
  }
  s = blockReduceSum(s);
  if (threadIdx.x == 0) {
    if (isX) {
      x2[row] = s;
      atomicAdd(out, s * (REG / ((float)BATCH * (float)FEAT)));
    } else {
      w2[row - BATCH] = s;
    }
  }
}

// --- Exact (f32) logit at the label class --------------------------------
__global__ void sgh_label_kernel(const float* __restrict__ X,
                                 const float* __restrict__ W,
                                 const int* __restrict__ y,
                                 const float* __restrict__ x2,
                                 const float* __restrict__ w2,
                                 float* __restrict__ lab) {
  const int m = blockIdx.x;
  const int c = y[m];
  const float* xr = X + (size_t)m * FEAT;
  const float* wr = W + (size_t)c * FEAT;
  const int k = threadIdx.x * 4;
  float4 a = *reinterpret_cast<const float4*>(xr + k);
  float4 b = *reinterpret_cast<const float4*>(wr + k);
  float s = a.x * b.x + a.y * b.y + a.z * b.z + a.w * b.w;
  s = blockReduceSum(s);
  if (threadIdx.x == 0)
    lab[m] = (2.0f * s - x2[m] - w2[c]) * (1.0f / (float)FEAT);
}

// --- Fused WMMA GEMM + online softmax -> lse[b] ---------------------------
// Block: 64 batch rows x all 8192 classes. 8 waves = 2 (mi, 32-row) x 4 (ni).
// Each wave: 32 rows x 32 classes per chunk = four 16x16 C tiles; class
// chunk = 128 columns across the 4 ni groups.
#define LDS_PAD 8
__global__ __launch_bounds__(256, 1) void sgh_gemm_lse_kernel(
    const float* __restrict__ X, const unsigned short* __restrict__ Wbf,
    const float* __restrict__ x2, const float* __restrict__ w2,
    float* __restrict__ lse) {
  __shared__ unsigned short sx[64][FEAT + LDS_PAD];  // X block as bf16 bits
  __shared__ float sred[64][4][2];                   // per-row {max,sum} per ni

  const int tid = threadIdx.x;
  const int lane = tid & 31;
  const int wv = tid >> 5;
  const int mi = wv & 1;   // 32-row half of the block
  const int ni = wv >> 1;  // 32-class column group
  const int lm = lane & 15;
  const int lh = lane >> 4;
  const int blockRow = blockIdx.x * 64;

  // Stage this block's X rows into LDS as bf16 (one-time).
  for (int i = tid * 4; i < 64 * FEAT; i += 256 * 4) {
    const int r = i >> 10;
    const int k = i & (FEAT - 1);
    float4 v = *reinterpret_cast<const float4*>(X + (size_t)(blockRow + r) * FEAT + k);
    *reinterpret_cast<ushort4*>(&sx[r][k]) =
        make_ushort4(f2b(v.x), f2b(v.y), f2b(v.z), f2b(v.w));
  }
  __syncthreads();

  // Per-lane squared norms for the 16 rows this lane owns in C tiles.
  float x2r[2][8];
#pragma unroll
  for (int t = 0; t < 2; ++t)
#pragma unroll
    for (int r = 0; r < 8; ++r)
      x2r[t][r] = x2[blockRow + mi * 32 + t * 16 + lh * 8 + r];

  float runMax[2][8], runSum[2][8];
#pragma unroll
  for (int t = 0; t < 2; ++t)
#pragma unroll
    for (int r = 0; r < 8; ++r) { runMax[t][r] = -1.0e30f; runSum[t][r] = 0.0f; }

  const float inv = 1.0f / (float)FEAT;
  const int arow0 = mi * 32 + lm;       // LDS row for A tile 0 (m = lane%16)
  const int arow1 = arow0 + 16;         // A tile 1

  for (int cb = 0; cb < NCLS; cb += 128) {
    const int wrow0 = cb + ni * 32 + lm;                // B tile 0: n = lane%16
    const int wrow1 = wrow0 + 16;                       // B tile 1
    const unsigned short* wp0 = Wbf + (size_t)wrow0 * FEAT;
    const unsigned short* wp1 = Wbf + (size_t)wrow1 * FEAT;

    v8f acc00 = {};  // [M-tile 0][N-tile 0]
    v8f acc01 = {};  // [M-tile 0][N-tile 1]
    v8f acc10 = {};  // [M-tile 1][N-tile 0]
    v8f acc11 = {};  // [M-tile 1][N-tile 1]
#pragma unroll 8
    for (int kb = 0; kb < FEAT; kb += 32) {
      // B 32x16 bf16: lane holds 16 contiguous K at kb + (lane/16)*16
      v16bf b0 = *reinterpret_cast<const v16bf*>(wp0 + kb + lh * 16);
      v16bf b1 = *reinterpret_cast<const v16bf*>(wp1 + kb + lh * 16);
      // A 16x32 bf16: lane holds 16 contiguous K at kb + (lane/16)*16
      union { v16bf v; v8bf h[2]; } a0, a1;
      a0.h[0] = *reinterpret_cast<const v8bf*>(&sx[arow0][kb + lh * 16]);
      a0.h[1] = *reinterpret_cast<const v8bf*>(&sx[arow0][kb + lh * 16 + 8]);
      a1.h[0] = *reinterpret_cast<const v8bf*>(&sx[arow1][kb + lh * 16]);
      a1.h[1] = *reinterpret_cast<const v8bf*>(&sx[arow1][kb + lh * 16 + 8]);
      acc00 = __builtin_amdgcn_wmma_f32_16x16x32_bf16(false, a0.v, false, b0,
                                                      (short)0, acc00, false, false);
      acc01 = __builtin_amdgcn_wmma_f32_16x16x32_bf16(false, a0.v, false, b1,
                                                      (short)0, acc01, false, false);
      acc10 = __builtin_amdgcn_wmma_f32_16x16x32_bf16(false, a1.v, false, b0,
                                                      (short)0, acc10, false, false);
      acc11 = __builtin_amdgcn_wmma_f32_16x16x32_bf16(false, a1.v, false, b1,
                                                      (short)0, acc11, false, false);
    }

    // logits for this 32x32 chunk + online softmax update (per C row).
    const float w2n0 = w2[wrow0];
    const float w2n1 = w2[wrow1];
#pragma unroll
    for (int t = 0; t < 2; ++t) {
      v8f& accA = t ? acc10 : acc00;
      v8f& accB = t ? acc11 : acc01;
#pragma unroll
      for (int r = 0; r < 8; ++r) {
        const float l0 = (2.0f * accA[r] - x2r[t][r] - w2n0) * inv;
        const float l1 = (2.0f * accB[r] - x2r[t][r] - w2n1) * inv;
        // row of C lives in one 16-lane half: butterfly over lane bits 0..3
        float v = fmaxf(l0, l1);
        v = fmaxf(v, __shfl_xor(v, 1, 32));
        v = fmaxf(v, __shfl_xor(v, 2, 32));
        v = fmaxf(v, __shfl_xor(v, 4, 32));
        v = fmaxf(v, __shfl_xor(v, 8, 32));
        float e = __expf(l0 - v) + __expf(l1 - v);
        e += __shfl_xor(e, 1, 32);
        e += __shfl_xor(e, 2, 32);
        e += __shfl_xor(e, 4, 32);
        e += __shfl_xor(e, 8, 32);
        const float M = fmaxf(runMax[t][r], v);
        runSum[t][r] = runSum[t][r] * __expf(runMax[t][r] - M) + e * __expf(v - M);
        runMax[t][r] = M;
      }
    }
  }

  // Combine the 4 ni-waves per row and emit lse.
  if (lm == 0) {
#pragma unroll
    for (int t = 0; t < 2; ++t)
#pragma unroll
      for (int r = 0; r < 8; ++r) {
        const int row = mi * 32 + t * 16 + lh * 8 + r;
        sred[row][ni][0] = runMax[t][r];
        sred[row][ni][1] = runSum[t][r];
      }
  }
  __syncthreads();
  if (tid < 64) {
    float M = sred[tid][0][0];
    M = fmaxf(M, sred[tid][1][0]);
    M = fmaxf(M, sred[tid][2][0]);
    M = fmaxf(M, sred[tid][3][0]);
    float S = 0.0f;
#pragma unroll
    for (int i = 0; i < 4; ++i)
      S += sred[tid][i][1] * __expf(sred[tid][i][0] - M);
    lse[blockRow + tid] = M + __logf(S);
  }
}

// --- Final: loss += mean(lse - logit_y) -----------------------------------
__global__ void sgh_finish_kernel(const float* __restrict__ lse,
                                  const float* __restrict__ lab,
                                  float* __restrict__ out) {
  float s = 0.0f;
  for (int i = blockIdx.x * blockDim.x + threadIdx.x; i < BATCH;
       i += gridDim.x * blockDim.x)
    s += lse[i] - lab[i];
  s = blockReduceSum(s);
  if (threadIdx.x == 0) atomicAdd(out, s * (1.0f / (float)BATCH));
}

extern "C" void kernel_launch(void* const* d_in, const int* in_sizes, int n_in,
                              void* d_out, int out_size, void* d_ws, size_t ws_size,
                              hipStream_t stream) {
  const float* X = (const float*)d_in[0];   // (8192, 1024) f32
  const float* W = (const float*)d_in[1];   // (8192, 1024) f32
  const int*   y = (const int*)d_in[2];     // (8192,) int
  float* out = (float*)d_out;               // scalar loss

  // Workspace layout: Wbf16 (16MB) | x2 | w2 | lab | lse  (~16.13 MB total)
  unsigned short* wbf = (unsigned short*)d_ws;
  float* x2  = (float*)((char*)d_ws + (size_t)NCLS * FEAT * sizeof(unsigned short));
  float* w2  = x2 + BATCH;
  float* lab = w2 + NCLS;
  float* lse = lab + BATCH;

  hipMemsetAsync(d_out, 0, sizeof(float), stream);
  sgh_prep_kernel<<<BATCH + NCLS, 256, 0, stream>>>(X, W, x2, w2, wbf, out);
  sgh_label_kernel<<<BATCH, 256, 0, stream>>>(X, W, y, x2, w2, lab);
  sgh_gemm_lse_kernel<<<BATCH / 64, 256, 0, stream>>>(X, wbf, x2, w2, lse);
  sgh_finish_kernel<<<16, 256, 0, stream>>>(lse, lab, out);
}
